// MiMoV2FlashDecoderLayer_7292854468632
// MI455X (gfx1250) — matrix operand order repaired and checked by reading, not compiled, not run on hardware
//
#include <hip/hip_runtime.h>

#define T_ 2048
#define H_ 2048
#define NH_ 16
#define NKV_ 4
#define HD_ 128
#define E_ 16
#define IM_ 1024
#define SCALE_ 0.08838834764831845f   // 128^-0.5

typedef unsigned short u16;
typedef __attribute__((ext_vector_type(8)))  float  v8f;
typedef __attribute__((ext_vector_type(16))) __bf16 v16b;

union AFrag { v16b v; unsigned u[8]; u16 s[16]; };

__device__ __forceinline__ u16 f2bf(float f) {
  unsigned x = __float_as_uint(f);
  unsigned r = (x + 0x7FFFu + ((x >> 16) & 1u)) >> 16;
  return (u16)r;
}

// bf16 pair pack (lo = k even, hi = k odd)
__device__ __forceinline__ unsigned pack2bf(float lo, float hi) {
  return (unsigned)f2bf(lo) | ((unsigned)f2bf(hi) << 16);
}

// ---- CDNA5 async global->LDS copy (ASYNCcnt-tracked, bypasses VGPRs) ----
__device__ __forceinline__ void async_g2l_b128(unsigned lds_off, const void* gaddr) {
  asm volatile("global_load_async_to_lds_b128 %0, %1, off"
               :: "v"(lds_off), "v"(gaddr) : "memory");
}
__device__ __forceinline__ void wait_async0() {
  asm volatile("s_wait_asynccnt 0" ::: "memory");
}
// LDS byte offset of a shared-memory pointer (flat LDS addr truncates to offset)
__device__ __forceinline__ unsigned lds_off_of(const void* p) {
  return (unsigned)(unsigned long long)p;
}

// ---------------------------------------------------------------------------
// Generic 128x128 tile bf16 WMMA GEMM core.
//  A: bf16 row-major (optionally row-gathered via gidx) -> LDS via
//     global_load_async_to_lds_b128 (no VGPR staging).
//  B: f32 row-major -> converted to bf16 pairs, staged K-transposed via DS.
//  256 threads = 8 waves; wave (wm 0..1, wn 0..3) owns 64x32 = 4x2 WMMA frags.
// LDS layouts (uint = packed bf16 pair along K):
//  au[row][pair]  row stride 20 uints (16 data + 4 pad; 16B-aligned rows)
//  bu[col][pair]  col stride 17 uints
// ---------------------------------------------------------------------------
__device__ __forceinline__ void gemm_core(
    const u16* __restrict__ A, int lda,
    const float* __restrict__ B, int ldb, int K,
    int m0, const int* gidx, int gcount, int n0,
    unsigned* au, unsigned* bu, v8f (&c)[4][2]) {
  const int tid  = threadIdx.x;
  const int lane = tid & 31;
  const int w    = tid >> 5;
  const int wm   = w >> 2, wn = w & 3;
  const int hlf  = lane >> 4, l15 = lane & 15;
  const unsigned abase = lds_off_of(au);

  for (int kb = 0; kb < K; kb += 32) {
    // ---- stage A tile: 128 rows x 32 bf16, async DMA to LDS ----
    {
      int row = tid >> 1;
      int seg = (tid & 1) * 32;                    // byte offset within row
      int m   = m0 + row;
      int src = gidx ? gidx[(m < gcount) ? m : 0] : m;
      const char* ga = (const char*)(A + (size_t)src * lda + kb) + seg;
      unsigned lo = abase + row * 80 + seg;
      async_g2l_b128(lo, ga);
      async_g2l_b128(lo + 16, ga + 16);
    }
    // ---- stage B tile: 32 x 128 f32 -> packed bf16 pairs, K-transposed ----
#pragma unroll
    for (int it = 0; it < 2; ++it) {
      int gl = tid + it * 256;                     // 0..511
      int kp = gl >> 5;                            // k-pair 0..15
      int c4 = gl & 31;                            // float4 col group
      const float* b0 = B + (size_t)(kb + 2 * kp) * ldb + n0 + c4 * 4;
      float4 f0 = *(const float4*)b0;
      float4 f1 = *(const float4*)(b0 + ldb);
      int nb0 = c4 * 4;
      bu[(nb0 + 0) * 17 + kp] = pack2bf(f0.x, f1.x);
      bu[(nb0 + 1) * 17 + kp] = pack2bf(f0.y, f1.y);
      bu[(nb0 + 2) * 17 + kp] = pack2bf(f0.z, f1.z);
      bu[(nb0 + 3) * 17 + kp] = pack2bf(f0.w, f1.w);
    }
    if (kb + 32 < K)  // pull next B tile toward L2 (global_prefetch_b8)
      __builtin_prefetch(B + (size_t)(kb + 32 + (tid >> 3)) * ldb + n0 + (tid & 7) * 16, 0, 1);
    wait_async0();
    __syncthreads();

    // ---- load fragments (ISA 16-bit A 16x32 / B 32x16 layouts) ----
    AFrag af[4], bfr[2];
#pragma unroll
    for (int fr = 0; fr < 4; ++fr) {
      const unsigned* base = au + (wm * 64 + fr * 16 + l15) * 20;
      int p0 = hlf ? 4 : 0;
#pragma unroll
      for (int i = 0; i < 4; ++i) { af[fr].u[i] = base[p0 + i]; af[fr].u[4 + i] = base[8 + p0 + i]; }
    }
#pragma unroll
    for (int fc = 0; fc < 2; ++fc) {
      const unsigned* base = bu + (wn * 32 + fc * 16 + l15) * 17 + (hlf ? 4 : 0);
#pragma unroll
      for (int i = 0; i < 8; ++i) bfr[fc].u[i] = base[i + (i & 4)];
    }
#pragma unroll
    for (int fr = 0; fr < 4; ++fr)
#pragma unroll
      for (int fc = 0; fc < 2; ++fc)
        c[fr][fc] = __builtin_amdgcn_wmma_f32_16x16x32_bf16(
            false, af[fr].v, false, bfr[fc].v, (short)0, c[fr][fc], false, false);
    __syncthreads();
  }
}

// ---------------------------------------------------------------------------
__global__ void k_zero(float* out, int n, int* ecnt) {
  int i = blockIdx.x * 256 + threadIdx.x;
  if (i < n) out[i] = 0.f;
  if (i < E_) ecnt[i] = 0;
}

// residual = hidden + residual; x1 = rmsnorm(residual) * w  (bf16 out)
__global__ __launch_bounds__(256) void k_addnorm(const float* __restrict__ hid,
                                                 const float* __restrict__ res,
                                                 const float* __restrict__ w,
                                                 float* res1, u16* xb) {
  __shared__ float red[256];
  __shared__ float rinv;
  int t = blockIdx.x;
  float vals[8];
  float ss = 0.f;
#pragma unroll
  for (int j = 0; j < 8; ++j) {
    int i = threadIdx.x + j * 256;
    float v = hid[(size_t)t * H_ + i] + res[(size_t)t * H_ + i];
    vals[j] = v; ss += v * v;
  }
  red[threadIdx.x] = ss; __syncthreads();
  for (int s = 128; s > 0; s >>= 1) { if (threadIdx.x < s) red[threadIdx.x] += red[threadIdx.x + s]; __syncthreads(); }
  if (threadIdx.x == 0) rinv = rsqrtf(red[0] * (1.f / H_) + 1e-6f);
  __syncthreads();
  float ri = rinv;
#pragma unroll
  for (int j = 0; j < 8; ++j) {
    int i = threadIdx.x + j * 256;
    res1[(size_t)t * H_ + i] = vals[j];
    xb[(size_t)t * H_ + i] = f2bf(vals[j] * ri * w[i]);
  }
}

// x2 = rmsnorm(res2) * w ; write bf16 + f32 copies
__global__ __launch_bounds__(256) void k_norm2(const float* __restrict__ r2,
                                               const float* __restrict__ w,
                                               u16* xb, float* xf) {
  __shared__ float red[256];
  __shared__ float rinv;
  int t = blockIdx.x;
  float vals[8];
  float ss = 0.f;
#pragma unroll
  for (int j = 0; j < 8; ++j) {
    int i = threadIdx.x + j * 256;
    float v = r2[(size_t)t * H_ + i];
    vals[j] = v; ss += v * v;
  }
  red[threadIdx.x] = ss; __syncthreads();
  for (int s = 128; s > 0; s >>= 1) { if (threadIdx.x < s) red[threadIdx.x] += red[threadIdx.x + s]; __syncthreads(); }
  if (threadIdx.x == 0) rinv = rsqrtf(red[0] * (1.f / H_) + 1e-6f);
  __syncthreads();
  float ri = rinv;
#pragma unroll
  for (int j = 0; j < 8; ++j) {
    int i = threadIdx.x + j * 256;
    float xn = vals[j] * ri * w[i];
    xb[(size_t)t * H_ + i] = f2bf(xn);
    xf[(size_t)t * H_ + i] = xn;
  }
}

// qkv f32 -> rope-rotated q/k bf16 + v bf16
__global__ __launch_bounds__(256) void k_rope(const float* __restrict__ qkv,
                                              const int* __restrict__ pos,
                                              u16* Qo, u16* Ko, u16* Vo) {
  int t = blockIdx.x;
  float p = (float)pos[t];
  for (int i = threadIdx.x; i < NH_ * 64; i += 256) {
    int hh = i >> 6, d = i & 63;
    float invf = __expf(-(float)d * (13.815510558f / 64.f));  // theta=1e6
    float ang = p * invf, sn, cs; __sincosf(ang, &sn, &cs);
    size_t b = (size_t)t * 3072 + hh * HD_ + d;
    float x1 = qkv[b], x2 = qkv[b + 64];
    size_t o = (size_t)t * 2048 + hh * HD_ + d;
    Qo[o]      = f2bf(x1 * cs - x2 * sn);
    Qo[o + 64] = f2bf(x2 * cs + x1 * sn);
  }
  for (int i = threadIdx.x; i < NKV_ * 64; i += 256) {
    int hh = i >> 6, d = i & 63;
    float invf = __expf(-(float)d * (13.815510558f / 64.f));
    float ang = p * invf, sn, cs; __sincosf(ang, &sn, &cs);
    size_t b = (size_t)t * 3072 + 2048 + hh * HD_ + d;
    float x1 = qkv[b], x2 = qkv[b + 64];
    size_t o = (size_t)t * 512 + hh * HD_ + d;
    Ko[o]      = f2bf(x1 * cs - x2 * sn);
    Ko[o + 64] = f2bf(x2 * cs + x1 * sn);
  }
  for (int i = threadIdx.x; i < 512; i += 256)
    Vo[(size_t)t * 512 + i] = f2bf(qkv[(size_t)t * 3072 + 2560 + i]);
}

// ---------------------------------------------------------------------------
// Flash attention: one wave per (16 query rows, head). WMMA for QK^T and PV.
// V tile staged to LDS via async DMA; P transposed through LDS.
// ---------------------------------------------------------------------------
__global__ __launch_bounds__(32) void k_attn(const u16* __restrict__ Q,
                                             const u16* __restrict__ Kk,
                                             const u16* __restrict__ V,
                                             u16* __restrict__ O) {
  __shared__ unsigned pu[16 * 17];   // P tile 16x32 bf16 (pair-packed along keys)
  __shared__ unsigned vu[32 * 68];   // V tile [key][dim-uint], 272B rows (16B aligned)
  int h = blockIdx.y, q0 = blockIdx.x * 16;
  int g = h >> 2;                    // NH/NKV = 4
  int lane = threadIdx.x, hlf = lane >> 4, l15 = lane & 15;
  const unsigned* Qu = (const unsigned*)Q;
  const unsigned* Ku = (const unsigned*)Kk;
  const unsigned* Vu = (const unsigned*)V;
  const unsigned vbase = lds_off_of(vu);

  AFrag qa[4];
  int qr = q0 + l15;
#pragma unroll
  for (int f = 0; f < 4; ++f) {
    int base2 = qr * 1024 + h * 64 + f * 16;
    int p0 = hlf ? 4 : 0;
#pragma unroll
    for (int i = 0; i < 4; ++i) { qa[f].u[i] = Qu[base2 + p0 + i]; qa[f].u[4 + i] = Qu[base2 + 8 + p0 + i]; }
  }

  v8f o[8];
  float mi[8], li[8];
#pragma unroll
  for (int i = 0; i < 8; ++i) {
    mi[i] = -1e30f; li[i] = 0.f;
#pragma unroll
    for (int j = 0; j < 8; ++j) o[i][j] = 0.f;
  }

  for (int j0 = 0; j0 <= q0 + 15; j0 += 32) {
    // stage V block [32 keys][128 dims] via async DMA (overlaps QK^T WMMAs)
#pragma unroll
    for (int it = 0; it < 16; ++it) {
      int gl = lane + it * 32;               // 0..511 b128 chunks
      int ky = gl >> 4, ch = (gl & 15) * 16; // key, byte offset in row
      int kg = j0 + ky; if (kg > T_ - 1) kg = T_ - 1;
      async_g2l_b128(vbase + ky * 272 + ch,
                     (const char*)(Vu + kg * 256 + g * 64) + ch);
    }

    v8f s0, s1;
#pragma unroll
    for (int j = 0; j < 8; ++j) { s0[j] = 0.f; s1[j] = 0.f; }
    int key0 = j0 + l15;        if (key0 > T_ - 1) key0 = T_ - 1;
    int key1 = j0 + 16 + l15;   if (key1 > T_ - 1) key1 = T_ - 1;
#pragma unroll
    for (int f = 0; f < 4; ++f) {
      AFrag b0, b1;
      int kb = f * 16 + (hlf ? 4 : 0);
      const unsigned* p0 = Ku + key0 * 256 + g * 64 + kb;
      const unsigned* p1 = Ku + key1 * 256 + g * 64 + kb;
#pragma unroll
      for (int i = 0; i < 8; ++i) { b0.u[i] = p0[i + (i & 4)]; b1.u[i] = p1[i + (i & 4)]; }
      s0 = __builtin_amdgcn_wmma_f32_16x16x32_bf16(false, qa[f].v, false, b0.v, (short)0, s0, false, false);
      s1 = __builtin_amdgcn_wmma_f32_16x16x32_bf16(false, qa[f].v, false, b1.v, (short)0, s1, false, false);
    }
    // online softmax over 32 keys
    float al[8];
    u16* ps = (u16*)pu;
#pragma unroll
    for (int i = 0; i < 8; ++i) {
      int row = q0 + i + hlf * 8;
      int ka = j0 + l15, kbk = ka + 16;
      float a0 = (ka  <= row) ? s0[i] * SCALE_ : -1e30f;
      float a1 = (kbk <= row) ? s1[i] * SCALE_ : -1e30f;
      float mx = fmaxf(a0, a1);
      for (int off = 1; off < 16; off <<= 1) mx = fmaxf(mx, __shfl_xor(mx, off, 32));
      float mn = fmaxf(mi[i], mx);
      float alpha = __expf(mi[i] - mn);
      float p0v = __expf(a0 - mn), p1v = __expf(a1 - mn);
      float rs = p0v + p1v;
      for (int off = 1; off < 16; off <<= 1) rs += __shfl_xor(rs, off, 32);
      li[i] = li[i] * alpha + rs;
      mi[i] = mn; al[i] = alpha;
      int rl = i + hlf * 8;
      ps[rl * 34 + l15]      = f2bf(p0v);
      ps[rl * 34 + 16 + l15] = f2bf(p1v);
    }
    wait_async0();
    __syncthreads();
    AFrag pa;
    {
      int p0 = hlf ? 4 : 0;
      const unsigned* pb = pu + l15 * 17;
#pragma unroll
      for (int i = 0; i < 4; ++i) { pa.u[i] = pb[p0 + i]; pa.u[4 + i] = pb[8 + p0 + i]; }
    }
    const u16* vs = (const u16*)vu;
#pragma unroll
    for (int nb = 0; nb < 8; ++nb) {
      AFrag vf;
      int dim = nb * 16 + l15;
      int kk0 = hlf ? 8 : 0;
#pragma unroll
      for (int i = 0; i < 8; ++i) {
        unsigned lo = vs[(kk0 + 2 * i + 2 * (i & 4))     * 136 + dim];
        unsigned hi = vs[(kk0 + 2 * i + 2 * (i & 4) + 1) * 136 + dim];
        vf.u[i] = lo | (hi << 16);
      }
      v8f oc = o[nb];
#pragma unroll
      for (int i = 0; i < 8; ++i) oc[i] *= al[i];
      o[nb] = __builtin_amdgcn_wmma_f32_16x16x32_bf16(false, pa.v, false, vf.v, (short)0, oc, false, false);
    }
    __syncthreads();
  }
#pragma unroll
  for (int nb = 0; nb < 8; ++nb)
#pragma unroll
    for (int i = 0; i < 8; ++i) {
      int row = q0 + i + hlf * 8;
      int dim = nb * 16 + l15;
      O[(size_t)row * 2048 + h * HD_ + dim] = f2bf(o[nb][i] / li[i]);
    }
}

// ---------------------------------------------------------------------------
#define GEMM_EPILOGUE_VARS                                  \
  int lane = threadIdx.x & 31, w = threadIdx.x >> 5;        \
  int wm = w >> 2, wn = w & 3, hlf = lane >> 4, l15 = lane & 15;

__global__ __launch_bounds__(256) void k_gemm_qkv(const u16* __restrict__ X,
                                                  const float* __restrict__ W,
                                                  float* __restrict__ Y) {
  __shared__ unsigned au[128 * 20], bu[128 * 17];
  v8f c[4][2];
  for (int a = 0; a < 4; ++a) for (int b = 0; b < 2; ++b) for (int i = 0; i < 8; ++i) c[a][b][i] = 0.f;
  int m0 = blockIdx.x * 128, n0 = blockIdx.y * 128;
  gemm_core(X, H_, W, 3072, H_, m0, nullptr, 0, n0, au, bu, c);
  GEMM_EPILOGUE_VARS
#pragma unroll
  for (int fr = 0; fr < 4; ++fr)
#pragma unroll
    for (int fc = 0; fc < 2; ++fc)
#pragma unroll
      for (int i = 0; i < 8; ++i) {
        int row = m0 + wm * 64 + fr * 16 + i + hlf * 8;
        int col = n0 + wn * 32 + fc * 16 + l15;
        Y[(size_t)row * 3072 + col] = c[fr][fc][i];
      }
}

__global__ __launch_bounds__(256) void k_gemm_wo(const u16* __restrict__ X,
                                                 const float* __restrict__ W,
                                                 float* __restrict__ R) {
  __shared__ unsigned au[128 * 20], bu[128 * 17];
  v8f c[4][2];
  for (int a = 0; a < 4; ++a) for (int b = 0; b < 2; ++b) for (int i = 0; i < 8; ++i) c[a][b][i] = 0.f;
  int m0 = blockIdx.x * 128, n0 = blockIdx.y * 128;
  gemm_core(X, H_, W, H_, H_, m0, nullptr, 0, n0, au, bu, c);
  GEMM_EPILOGUE_VARS
#pragma unroll
  for (int fr = 0; fr < 4; ++fr)
#pragma unroll
    for (int fc = 0; fc < 2; ++fc)
#pragma unroll
      for (int i = 0; i < 8; ++i) {
        int row = m0 + wm * 64 + fr * 16 + i + hlf * 8;
        int col = n0 + wn * 32 + fc * 16 + l15;
        size_t idx = (size_t)row * H_ + col;
        R[idx] = R[idx] + c[fr][fc][i];   // residual2 = attn@wo + residual1
      }
}

// sigmoid gating + group-top2 / top-4 expert routing with per-expert compaction
__global__ __launch_bounds__(256) void k_gate(const float* __restrict__ x2f,
                                              const float* __restrict__ gw,
                                              const float* __restrict__ gb,
                                              int* ecnt, int* etok, float* ewgt) {
  __shared__ float lg[E_];
  int t = blockIdx.x;
  int e = threadIdx.x >> 4, sub = threadIdx.x & 15;
  float acc = 0.f;
  for (int i = sub; i < H_; i += 16) acc += x2f[(size_t)t * H_ + i] * gw[(size_t)e * H_ + i];
  for (int off = 8; off; off >>= 1) acc += __shfl_xor(acc, off, 32);
  if (sub == 0) lg[e] = acc;
  __syncthreads();
  if (threadIdx.x == 0) {
    float s[E_], sc[E_];
    for (int i = 0; i < E_; ++i) { s[i] = 1.f / (1.f + __expf(-lg[i])); sc[i] = s[i] + gb[i]; }
    float gs[4];
    for (int gI = 0; gI < 4; ++gI) {
      float m1 = -1e30f, m2 = -1e30f;
      for (int j = 0; j < 4; ++j) {
        float v = sc[gI * 4 + j];
        if (v > m1) { m2 = m1; m1 = v; } else if (v > m2) m2 = v;
      }
      gs[gI] = m1 + m2;
    }
    int g1 = 0; for (int gI = 1; gI < 4; ++gI) if (gs[gI] > gs[g1]) g1 = gI;
    int g2 = -1; for (int gI = 0; gI < 4; ++gI) if (gI != g1 && (g2 < 0 || gs[gI] > gs[g2])) g2 = gI;
    float msk[E_];
    for (int i = 0; i < E_; ++i) { int gg = i >> 2; msk[i] = (gg == g1 || gg == g2) ? sc[i] : -1e30f; }
    int sel[4]; float wsum = 0.f;
    for (int k = 0; k < 4; ++k) {
      int bi = 0; float bv = -1e31f;
      for (int i = 0; i < E_; ++i) if (msk[i] > bv) { bv = msk[i]; bi = i; }
      sel[k] = bi; msk[bi] = -3e38f; wsum += s[bi];
    }
    for (int k = 0; k < 4; ++k) {
      int ee = sel[k];
      int pos = atomicAdd(&ecnt[ee], 1);
      etok[ee * T_ + pos] = t;
      ewgt[ee * T_ + pos] = s[ee] / wsum;
    }
  }
}

__global__ __launch_bounds__(256) void k_gemm_gu(const u16* __restrict__ X,
                                                 const float* __restrict__ W,
                                                 const int* __restrict__ etok,
                                                 const int* __restrict__ cnt,
                                                 float* __restrict__ GU) {
  int cn = *cnt;
  int m0 = blockIdx.x * 128;
  if (m0 >= cn) return;
  __shared__ unsigned au[128 * 20], bu[128 * 17];
  v8f c[4][2];
  for (int a = 0; a < 4; ++a) for (int b = 0; b < 2; ++b) for (int i = 0; i < 8; ++i) c[a][b][i] = 0.f;
  int n0 = blockIdx.y * 128;
  gemm_core(X, H_, W, 2 * IM_, H_, m0, etok, cn, n0, au, bu, c);
  GEMM_EPILOGUE_VARS
#pragma unroll
  for (int fr = 0; fr < 4; ++fr)
#pragma unroll
    for (int fc = 0; fc < 2; ++fc)
#pragma unroll
      for (int i = 0; i < 8; ++i) {
        int row = m0 + wm * 64 + fr * 16 + i + hlf * 8;
        int col = n0 + wn * 32 + fc * 16 + l15;
        if (row < cn) GU[(size_t)row * (2 * IM_) + col] = c[fr][fc][i];
      }
}

__global__ __launch_bounds__(256) void k_silu(const float* __restrict__ gu,
                                              const int* __restrict__ cnt,
                                              u16* __restrict__ hb) {
  int i = blockIdx.x * 256 + threadIdx.x;
  int row = i >> 10, c = i & (IM_ - 1);
  if (row >= *cnt) return;
  float g = gu[(size_t)row * (2 * IM_) + c];
  float u = gu[(size_t)row * (2 * IM_) + IM_ + c];
  float sg = g / (1.f + __expf(-g));
  hb[(size_t)row * IM_ + c] = f2bf(sg * u);
}

__global__ __launch_bounds__(256) void k_gemm_down(const u16* __restrict__ Hh,
                                                   const float* __restrict__ W,
                                                   const int* __restrict__ etok,
                                                   const float* __restrict__ ewgt,
                                                   const int* __restrict__ cnt,
                                                   float* __restrict__ Out) {
  int cn = *cnt;
  int m0 = blockIdx.x * 128;
  if (m0 >= cn) return;
  __shared__ unsigned au[128 * 20], bu[128 * 17];
  v8f c[4][2];
  for (int a = 0; a < 4; ++a) for (int b = 0; b < 2; ++b) for (int i = 0; i < 8; ++i) c[a][b][i] = 0.f;
  int n0 = blockIdx.y * 128;
  gemm_core(Hh, IM_, W, H_, IM_, m0, nullptr, 0, n0, au, bu, c);
  GEMM_EPILOGUE_VARS
#pragma unroll
  for (int fr = 0; fr < 4; ++fr)
#pragma unroll
    for (int fc = 0; fc < 2; ++fc)
#pragma unroll
      for (int i = 0; i < 8; ++i) {
        int row = m0 + wm * 64 + fr * 16 + i + hlf * 8;
        int col = n0 + wn * 32 + fc * 16 + l15;
        if (row < cn) {
          int t = etok[row];
          float wt = ewgt[row];
          Out[(size_t)t * H_ + col] += wt * c[fr][fc][i];  // experts stream-serialized
        }
      }
}

// ---------------------------------------------------------------------------
extern "C" void kernel_launch(void* const* d_in, const int* in_sizes, int n_in,
                              void* d_out, int out_size, void* d_ws, size_t ws_size,
                              hipStream_t stream) {
  (void)in_sizes; (void)n_in; (void)out_size; (void)ws_size;
  const int*   positions = (const int*)d_in[0];
  const float* hidden    = (const float*)d_in[1];
  const float* resid     = (const float*)d_in[2];
  const float* wln1      = (const float*)d_in[3];
  const float* wln2      = (const float*)d_in[4];
  const float* wqkv      = (const float*)d_in[5];
  const float* wo        = (const float*)d_in[6];
  const float* gatew     = (const float*)d_in[7];
  const float* gateb     = (const float*)d_in[8];
  const float* wgu       = (const float*)d_in[9];
  const float* wdn       = (const float*)d_in[10];

  float* out  = (float*)d_out;                 // [T,H] moe output
  float* res2 = out + (size_t)T_ * H_;         // [T,H] residual output (res1 then res2)

  char* ws = (char*)d_ws;
  size_t off = 0;
  auto alloc = [&](size_t bytes) -> void* {
    off = (off + 255) & ~(size_t)255;
    void* p = ws + off; off += bytes; return p;
  };
  u16*   x1b  = (u16*)  alloc((size_t)T_ * H_ * 2);
  float* qkv  = (float*)alloc((size_t)T_ * 3072 * 4);
  u16*   qb   = (u16*)  alloc((size_t)T_ * 2048 * 2);
  u16*   kb   = (u16*)  alloc((size_t)T_ * 512 * 2);
  u16*   vb   = (u16*)  alloc((size_t)T_ * 512 * 2);
  u16*   attn = (u16*)  alloc((size_t)T_ * 2048 * 2);
  u16*   x2b  = (u16*)  alloc((size_t)T_ * H_ * 2);
  float* x2f  = (float*)alloc((size_t)T_ * H_ * 4);
  float* gu   = (float*)alloc((size_t)T_ * 2 * IM_ * 4);
  u16*   hb   = (u16*)  alloc((size_t)T_ * IM_ * 2);
  int*   ecnt = (int*)  alloc(E_ * 4);
  int*   etok = (int*)  alloc((size_t)E_ * T_ * 4);
  float* ewgt = (float*)alloc((size_t)E_ * T_ * 4);

  k_zero<<<(T_ * H_ + 255) / 256, 256, 0, stream>>>(out, T_ * H_, ecnt);
  k_addnorm<<<T_, 256, 0, stream>>>(hidden, resid, wln1, res2, x1b);
  k_gemm_qkv<<<dim3(T_ / 128, 3072 / 128), 256, 0, stream>>>(x1b, wqkv, qkv);
  k_rope<<<T_, 256, 0, stream>>>(qkv, positions, qb, kb, vb);
  k_attn<<<dim3(T_ / 16, NH_), 32, 0, stream>>>(qb, kb, vb, attn);
  k_gemm_wo<<<dim3(T_ / 128, H_ / 128), 256, 0, stream>>>(attn, wo, res2);
  k_norm2<<<T_, 256, 0, stream>>>(res2, wln2, x2b, x2f);
  k_gate<<<T_, 256, 0, stream>>>(x2f, gatew, gateb, ecnt, etok, ewgt);
  for (int e = 0; e < E_; ++e) {
    k_gemm_gu<<<dim3(T_ / 128, (2 * IM_) / 128), 256, 0, stream>>>(
        x2b, wgu + (size_t)e * H_ * 2 * IM_, etok + e * T_, ecnt + e, gu);
    k_silu<<<(T_ * IM_) / 256, 256, 0, stream>>>(gu, ecnt + e, hb);
    k_gemm_down<<<dim3(T_ / 128, H_ / 128), 256, 0, stream>>>(
        hb, wdn + (size_t)e * IM_ * H_, etok + e * T_, ewgt + e * T_, ecnt + e, out);
  }
}
